// FoldingEncoder_2714419331072
// MI455X (gfx1250) — matrix operand order repaired and checked by reading, not compiled, not run on hardware
//
#include <hip/hip_runtime.h>
#include <hip/hip_bf16.h>
#include <math.h>

typedef __attribute__((ext_vector_type(16))) _Float16 v16h;
typedef __attribute__((ext_vector_type(8)))  _Float16 v8h;
typedef __attribute__((ext_vector_type(8)))  float    v8f;

#define BB 8
#define NN 2048
#define KNB 16      // neighbors
#define EPSV 1e-5f

// ---------------------------------------------------------------------------
// KNN: one wave (32 lanes) per query point, 8 queries per 256-thread block.
// ---------------------------------------------------------------------------
__global__ __launch_bounds__(256) void knn_kernel(const float* __restrict__ x,
                                                  const int* __restrict__ mask,
                                                  int* __restrict__ idx_out) {
    __shared__ float spx[NN], spy[NN], spz[NN], spen[NN];
    __shared__ float sd[8 * 32 * KNB];
    __shared__ int   si[8 * 32 * KNB];

    const int b     = blockIdx.x / (NN / 8);
    const int qbase = (blockIdx.x % (NN / 8)) * 8;
    const int tid   = threadIdx.x;
    const float* xb = x + b * 3 * NN;

    for (int i = tid; i < NN; i += 256) {
        spx[i]  = xb[i];
        spy[i]  = xb[NN + i];
        spz[i]  = xb[2 * NN + i];
        spen[i] = (mask[b * NN + i] == 1) ? 0.0f : __builtin_inff();
    }
    __syncthreads();

    const int wave = tid >> 5, lane = tid & 31;
    const int q = qbase + wave;
    const float qx = spx[q], qy = spy[q], qz = spz[q];

    float bd[KNB];
    int   bi[KNB];
#pragma unroll
    for (int j = 0; j < KNB; ++j) { bd[j] = __builtin_inff(); bi[j] = -1; }

    for (int m = lane; m < NN; m += 32) {
        float dx = spx[m] - qx, dy = spy[m] - qy, dz = spz[m] - qz;
        float d = dx * dx + dy * dy + dz * dz + spen[m];
        if (d < bd[KNB - 1]) {
            bd[KNB - 1] = d; bi[KNB - 1] = m;
#pragma unroll
            for (int j = KNB - 1; j > 0; --j) {
                if (bd[j] < bd[j - 1]) {
                    float td = bd[j]; bd[j] = bd[j - 1]; bd[j - 1] = td;
                    int   ti = bi[j]; bi[j] = bi[j - 1]; bi[j - 1] = ti;
                }
            }
        }
    }

    float* myd = sd + (wave * 32 + lane) * KNB;
    int*   myi = si + (wave * 32 + lane) * KNB;
#pragma unroll
    for (int j = 0; j < KNB; ++j) { myd[j] = bd[j]; myi[j] = bi[j]; }
    __syncthreads();

    int* out = idx_out + (b * NN + q) * KNB;
    int ptr = 0;
    for (int r = 0; r < KNB; ++r) {
        float v  = myd[ptr];
        int   hi = myi[ptr];
        int   wl = lane;
#pragma unroll
        for (int off = 16; off > 0; off >>= 1) {
            float ov = __shfl_xor(v, off, 32);
            int   ow = __shfl_xor(wl, off, 32);
            if (ov < v) { v = ov; wl = ow; }
        }
        int widx = __shfl(hi, wl, 32);
        if (lane == 0) out[r] = widx;
        if (lane == wl) ptr++;
    }
}

// ---------------------------------------------------------------------------
// Per-point neighbor covariance feature: [p(3), cov(9)] padded to 32 f16.
// ---------------------------------------------------------------------------
__global__ __launch_bounds__(256) void cov_feat_kernel(const float* __restrict__ x,
                                                       const int* __restrict__ idx,
                                                       _Float16* __restrict__ A0) {
    int gid = blockIdx.x * 256 + threadIdx.x;
    if (gid >= BB * NN) return;
    int b = gid / NN, n = gid % NN;
    const float* xb = x + b * 3 * NN;
    const int* id = idx + gid * KNB;

    float px[KNB], py[KNB], pz[KNB];
    float mx = 0.f, my = 0.f, mz = 0.f;
#pragma unroll
    for (int k = 0; k < KNB; ++k) {
        int j = id[k]; if (j < 0) j = n;
        px[k] = xb[j]; py[k] = xb[NN + j]; pz[k] = xb[2 * NN + j];
        mx += px[k]; my += py[k]; mz += pz[k];
    }
    mx *= (1.f / KNB); my *= (1.f / KNB); mz *= (1.f / KNB);
    float c00 = 0, c01 = 0, c02 = 0, c11 = 0, c12 = 0, c22 = 0;
#pragma unroll
    for (int k = 0; k < KNB; ++k) {
        float dx = px[k] - mx, dy = py[k] - my, dz = pz[k] - mz;
        c00 += dx * dx; c01 += dx * dy; c02 += dx * dz;
        c11 += dy * dy; c12 += dy * dz; c22 += dz * dz;
    }
    _Float16* o = A0 + gid * 32;
    o[0]  = (_Float16)xb[n]; o[1] = (_Float16)xb[NN + n]; o[2] = (_Float16)xb[2 * NN + n];
    o[3]  = (_Float16)c00; o[4]  = (_Float16)c01; o[5]  = (_Float16)c02;
    o[6]  = (_Float16)c01; o[7]  = (_Float16)c11; o[8]  = (_Float16)c12;
    o[9]  = (_Float16)c02; o[10] = (_Float16)c12; o[11] = (_Float16)c22;
#pragma unroll
    for (int k = 12; k < 32; ++k) o[k] = (_Float16)0.f;
}

// ---------------------------------------------------------------------------
// Pack f32 weights [Kreal, Nc] into fragment-ready f16:
//   Wp[ ((kb*2 + h)*Nc + col)*16 + 2v + p ] = W[kb*32 + 16h + 2v + p][col]
// so one lane's whole B fragment is 16 contiguous halves (32B load).
// Zero-pads K to Kpad.
// ---------------------------------------------------------------------------
__global__ void pack_w_kernel(const float* __restrict__ W, _Float16* __restrict__ Wp,
                              int Kreal, int Kpad, int Nc) {
    int i = blockIdx.x * 256 + threadIdx.x;           // one thread per dword (half pair)
    int total = (Kpad >> 5) * 2 * Nc * 8;
    if (i >= total) return;
    int v   = i & 7;
    int col = (i >> 3) % Nc;
    int hb  = (i >> 3) / Nc;                          // kb*2 + h
    int kb  = hb >> 1, h = hb & 1;
    int k   = (kb << 5) + 16 * h + 2 * v;
    float w0 = (k     < Kreal) ? W[k * Nc + col]       : 0.f;
    float w1 = (k + 1 < Kreal) ? W[(k + 1) * Nc + col] : 0.f;
    Wp[i * 2]     = (_Float16)w0;
    Wp[i * 2 + 1] = (_Float16)w1;
}

// ---------------------------------------------------------------------------
// WMMA GEMM + bias + BN stats. One wave computes a 64x16 output strip
// (4 M-tiles sharing each B fragment -> 4 v_wmma per K step).
// A: f16 row-major [M, Kpad]; Wp: packed f16 (see pack_w_kernel).
// ---------------------------------------------------------------------------
__global__ __launch_bounds__(256) void gemm_bn_kernel(const _Float16* __restrict__ A,
                                                      const _Float16* __restrict__ Wp,
                                                      const float* __restrict__ bias,
                                                      float* __restrict__ Y,
                                                      float* __restrict__ stats,
                                                      int M, int Kpad, int Nc) {
    const int wave = threadIdx.x >> 5, lane = threadIdx.x & 31;
    const int tilesN = Nc >> 4;
    const int tileId = blockIdx.x * 8 + wave;
    const int tmBase = (tileId / tilesN) * 64;
    const int tn     = tileId % tilesN;
    if (tmBase >= M) return;

    const int colB = tn * 16 + (lane & 15);
    const int hA   = (lane >= 16) ? 8 : 0;      // C/D: upper half-wave holds rows 8..15
    const int h    = (lane >= 16) ? 1 : 0;      // B: upper half-wave holds K=16..31
    const unsigned abase = (unsigned)(tmBase + (lane & 15)) * (unsigned)Kpad;
    const unsigned tstep = (unsigned)(16 * Kpad);

    v8f acc[4] = {v8f{}, v8f{}, v8f{}, v8f{}};
    const int kbCnt = Kpad >> 5;

    for (int kb = 0; kb < kbCnt; ++kb) {
        const int kk = kb << 5;
        const v16h bf = *(const v16h*)(Wp + ((unsigned)((kb * 2 + h) * Nc + colB) << 4));
        const _Float16* ap0 = A + abase + (unsigned)(kk + 16 * h);
        if (kb + 1 < kbCnt) {
            __builtin_prefetch(ap0 + 32, 0, 3);                 // global_prefetch_b8
            __builtin_prefetch(Wp + ((unsigned)(((kb + 1) * 2 + h) * Nc + colB) << 4), 0, 3);
        }
#pragma unroll
        for (int t = 0; t < 4; ++t) {
            const _Float16* ap = ap0 + t * tstep;
            v8h lo = *(const v8h*)(ap);
            v8h hi = *(const v8h*)(ap + 8);
            v16h af = __builtin_shufflevector(lo, hi, 0, 1, 2, 3, 4, 5, 6, 7,
                                              8, 9, 10, 11, 12, 13, 14, 15);
            acc[t] = __builtin_amdgcn_wmma_f32_16x16x32_f16(false, af, false, bf,
                                                            (short)0, acc[t], false, false);
        }
    }

    const float bval = bias[colB];
    float s = 0.f, s2 = 0.f;
#pragma unroll
    for (int t = 0; t < 4; ++t) {
#pragma unroll
        for (int r = 0; r < 8; ++r) {
            int row = tmBase + t * 16 + r + hA;     // C/D: VGPR r -> rows r / r+8
            float val = acc[t][r] + bval;
            Y[(unsigned)row * (unsigned)Nc + (unsigned)colB] = val;
            s += val; s2 += val * val;
        }
    }
    s  += __shfl_xor(s, 16, 32);
    s2 += __shfl_xor(s2, 16, 32);
    if (lane < 16) {
        atomicAdd(&stats[colB], s);
        atomicAdd(&stats[Nc + colB], s2);
    }
}

__global__ void zero_kernel(float* p, int n) {
    int i = blockIdx.x * 256 + threadIdx.x;
    if (i < n) p[i] = 0.f;
}

__global__ void bn_finalize_kernel(const float* __restrict__ stats,
                                   const float* __restrict__ gamma,
                                   const float* __restrict__ beta,
                                   float* __restrict__ params, int Nc, float invM) {
    int c = blockIdx.x * 256 + threadIdx.x;
    if (c >= Nc) return;
    float mean = stats[c] * invM;
    float var  = stats[Nc + c] * invM - mean * mean;
    float sc   = gamma[c] * rsqrtf(var + EPSV);
    params[c]      = sc;
    params[Nc + c] = beta[c] - mean * sc;
}

__global__ void bn_relu_kernel(const float* __restrict__ Y,
                               const float* __restrict__ params,
                               _Float16* __restrict__ H, int M, int Nc) {
    int i = blockIdx.x * 256 + threadIdx.x;
    if (i >= M * Nc) return;
    int c = i & (Nc - 1);                       // Nc is a power of two here
    float v = Y[i] * params[c] + params[Nc + c];
    H[i] = (_Float16)(v > 0.f ? v : 0.f);
}

__global__ void gather_max_kernel(const _Float16* __restrict__ H,
                                  const int* __restrict__ idx,
                                  _Float16* __restrict__ Out, int Nc) {
    int i = blockIdx.x * 256 + threadIdx.x;
    if (i >= BB * NN * Nc) return;
    int c = i & (Nc - 1), pt = i / Nc, b = pt / NN;
    const int* id = idx + pt * KNB;
    float m = -__builtin_inff();
#pragma unroll
    for (int k = 0; k < KNB; ++k) {
        int j = id[k];
        if (j < 0) continue;
        float v = (float)H[(unsigned)(b * NN + j) * (unsigned)Nc + (unsigned)c];
        m = fmaxf(m, v);
    }
    Out[i] = (_Float16)m;
}

__global__ __launch_bounds__(256) void segment_pool_kernel(const _Float16* __restrict__ H,
                                                           const int* __restrict__ mask,
                                                           float* __restrict__ pooled, int Nc) {
    __shared__ float red[256];
    int b = blockIdx.x / Nc, c = blockIdx.x % Nc;
    float m = -__builtin_inff();
    for (int n = threadIdx.x; n < NN; n += 256)
        if (mask[b * NN + n] == 1)
            m = fmaxf(m, (float)H[(unsigned)(b * NN + n) * (unsigned)Nc + (unsigned)c]);
    red[threadIdx.x] = m;
    __syncthreads();
    for (int s = 128; s > 0; s >>= 1) {
        if (threadIdx.x < s) red[threadIdx.x] = fmaxf(red[threadIdx.x], red[threadIdx.x + s]);
        __syncthreads();
    }
    if (threadIdx.x == 0) pooled[b * Nc + c] = red[0];
}

__global__ void small_gemm_kernel(const float* __restrict__ P, const float* __restrict__ W,
                                  const float* __restrict__ bias, float* __restrict__ Y,
                                  int K, int Nc) {
    int i = blockIdx.x * 256 + threadIdx.x;
    if (i >= BB * Nc) return;
    int m = i / Nc, c = i % Nc;
    float acc = bias[c];
    for (int k = 0; k < K; ++k) acc += P[m * K + k] * W[k * Nc + c];
    Y[i] = acc;
}

__global__ void small_bn_relu_kernel(const float* __restrict__ Y,
                                     const float* __restrict__ gamma,
                                     const float* __restrict__ beta,
                                     float* __restrict__ Out, int Nc) {
    int c = blockIdx.x * 256 + threadIdx.x;
    if (c >= Nc) return;
    float s = 0.f, s2 = 0.f;
    for (int m = 0; m < BB; ++m) { float v = Y[m * Nc + c]; s += v; s2 += v * v; }
    float mean = s * (1.f / BB);
    float var  = s2 * (1.f / BB) - mean * mean;
    float sc = gamma[c] * rsqrtf(var + EPSV);
    float sh = beta[c] - mean * sc;
    for (int m = 0; m < BB; ++m) {
        float v = Y[m * Nc + c] * sc + sh;
        Out[m * Nc + c] = v > 0.f ? v : 0.f;
    }
}

// ---------------------------------------------------------------------------
extern "C" void kernel_launch(void* const* d_in, const int* in_sizes, int n_in,
                              void* d_out, int out_size, void* d_ws, size_t ws_size,
                              hipStream_t stream) {
    const float* x    = (const float*)d_in[0];
    const int*   mask = (const int*)d_in[1];
    const float *w_m1a = (const float*)d_in[2],  *b_m1a = (const float*)d_in[3],
                *g_m1a = (const float*)d_in[4],  *be_m1a = (const float*)d_in[5];
    const float *w_m1b = (const float*)d_in[6],  *b_m1b = (const float*)d_in[7],
                *g_m1b = (const float*)d_in[8],  *be_m1b = (const float*)d_in[9];
    const float *w_m1c = (const float*)d_in[10], *b_m1c = (const float*)d_in[11],
                *g_m1c = (const float*)d_in[12], *be_m1c = (const float*)d_in[13];
    const float *w_gl1 = (const float*)d_in[14], *b_gl1 = (const float*)d_in[15],
                *g_gl1 = (const float*)d_in[16], *be_gl1 = (const float*)d_in[17];
    const float *w_gl2 = (const float*)d_in[18], *b_gl2 = (const float*)d_in[19],
                *g_gl2 = (const float*)d_in[20], *be_gl2 = (const float*)d_in[21];
    const float *w_m2a = (const float*)d_in[22], *b_m2a = (const float*)d_in[23],
                *g_m2a = (const float*)d_in[24], *be_m2a = (const float*)d_in[25];
    const float *w_m2b = (const float*)d_in[26], *b_m2b = (const float*)d_in[27],
                *g_m2b = (const float*)d_in[28], *be_m2b = (const float*)d_in[29];

    char* ws = (char*)d_ws;
    size_t off = 0;
    auto take = [&](size_t bytes) -> void* {
        void* p = ws + off;
        off += (bytes + 255) & ~(size_t)255;
        return p;
    };
    const int M = BB * NN;                              // 16384 rows
    int*      idx    = (int*)take((size_t)M * KNB * 4);
    _Float16* A0     = (_Float16*)take((size_t)M * 32 * 2);
    float*    Ybuf   = (float*)take((size_t)M * 1024 * 4);
    _Float16* Ha     = (_Float16*)take((size_t)M * 1024 * 2);
    _Float16* Hb     = (_Float16*)take((size_t)M * 1024 * 2);
    _Float16* Wp     = (_Float16*)take((size_t)128 * 1024 * 2);  // max packed weights
    float*    stats  = (float*)take(2 * 1024 * 4);
    float*    params = (float*)take(2 * 1024 * 4);
    float*    pooled = (float*)take(BB * 1024 * 4);
    float*    s1     = (float*)take(BB * 1024 * 4);
    float*    s2     = (float*)take(BB * 1024 * 4);

    const float invM = 1.f / (float)M;

    auto layer = [&](const _Float16* Ain, int Kreal, int Kpad, int Nc,
                     const float* W, const float* bias, const float* gamma,
                     const float* beta, _Float16* Hout) {
        int packN = (Kpad >> 5) * 2 * Nc * 8;
        pack_w_kernel<<<(packN + 255) / 256, 256, 0, stream>>>(W, Wp, Kreal, Kpad, Nc);
        zero_kernel<<<(2 * Nc + 255) / 256, 256, 0, stream>>>(stats, 2 * Nc);
        int tiles = (M / 64) * (Nc / 16);
        gemm_bn_kernel<<<(tiles + 7) / 8, 256, 0, stream>>>(Ain, Wp, bias, Ybuf, stats,
                                                            M, Kpad, Nc);
        bn_finalize_kernel<<<(Nc + 255) / 256, 256, 0, stream>>>(stats, gamma, beta,
                                                                 params, Nc, invM);
        bn_relu_kernel<<<(M * Nc + 255) / 256, 256, 0, stream>>>(Ybuf, params, Hout, M, Nc);
    };

    // 1) KNN + covariance features
    knn_kernel<<<BB * (NN / 8), 256, 0, stream>>>(x, mask, idx);
    cov_feat_kernel<<<(M + 255) / 256, 256, 0, stream>>>(x, idx, A0);

    // 2) point MLP 12->64->64->64
    layer(A0, 12, 32, 64, w_m1a, b_m1a, g_m1a, be_m1a, Ha);
    layer(Ha, 64, 64, 64, w_m1b, b_m1b, g_m1b, be_m1b, Hb);
    layer(Hb, 64, 64, 64, w_m1c, b_m1c, g_m1c, be_m1c, Ha);

    // 3) gather-max + 64->128
    gather_max_kernel<<<(M * 64 + 255) / 256, 256, 0, stream>>>(Ha, idx, Hb, 64);
    layer(Hb, 64, 64, 128, w_gl1, b_gl1, g_gl1, be_gl1, Ha);

    // 4) gather-max + 128->1024
    gather_max_kernel<<<(M * 128 + 255) / 256, 256, 0, stream>>>(Ha, idx, Hb, 128);
    layer(Hb, 128, 128, 1024, w_gl2, b_gl2, g_gl2, be_gl2, Ha);

    // 5) masked segment max -> (8, 1024)
    segment_pool_kernel<<<BB * 1024, 256, 0, stream>>>(Ha, mask, pooled, 1024);

    // 6) tail MLP on 8 rows: 1024->512->512
    small_gemm_kernel<<<(BB * 512 + 255) / 256, 256, 0, stream>>>(pooled, w_m2a, b_m2a,
                                                                  s1, 1024, 512);
    small_bn_relu_kernel<<<(512 + 255) / 256, 256, 0, stream>>>(s1, g_m2a, be_m2a, s2, 512);
    small_gemm_kernel<<<(BB * 512 + 255) / 256, 256, 0, stream>>>(s2, w_m2b, b_m2b,
                                                                  s1, 512, 512);
    small_bn_relu_kernel<<<(512 + 255) / 256, 256, 0, stream>>>(s1, g_m2b, be_m2b,
                                                                (float*)d_out, 512);
}
